// MultiheadSelfAttention_70617852281444
// MI455X (gfx1250) — compile-verified, matrix operand order checked by reading, not run on hardware
//
#include <hip/hip_runtime.h>
#include <hip/hip_bf16.h>
#include <cstdint>
#include <cstddef>

#define B_DIM 2
#define T_DIM 2048
#define M_DIM 2048
#define H_DIM 16
#define DHEAD 128
#define NEG_INF (-1.0e30f)

// CDNA5 async global->LDS path (ASYNCcnt), with safe fallback.
#if defined(__has_builtin)
# if __has_builtin(__builtin_amdgcn_global_load_async_to_lds_b128) && \
     __has_builtin(__builtin_amdgcn_s_wait_asynccnt)
#  define HAVE_ASYNC_LDS 1
# endif
#endif
#ifndef HAVE_ASYNC_LDS
# define HAVE_ASYNC_LDS 0
#endif

typedef __attribute__((ext_vector_type(16))) __bf16 v16bf;
typedef __attribute__((ext_vector_type(8)))  float  v8f;
typedef __attribute__((ext_vector_type(4)))  int    vi4;

#if HAVE_ASYNC_LDS
// Builtin signature (from hipcc diagnostic): pointee is a 4 x i32 vector,
// global source in addrspace(1), LDS destination in addrspace(3).
typedef __attribute__((address_space(1))) vi4 gvi4;
typedef __attribute__((address_space(3))) vi4 lvi4;
#endif

union Frag {
  v16bf bf;
  uint4 u[2];
};

__device__ __forceinline__ unsigned short f2bf(float f) {
  unsigned int u = __float_as_uint(f);
  u += 0x7fffu + ((u >> 16) & 1u);   // round-to-nearest-even
  return (unsigned short)(u >> 16);
}

__device__ __forceinline__ v8f wmma_bf16(const Frag& a, const Frag& b, v8f c) {
  return __builtin_amdgcn_wmma_f32_16x16x32_bf16(false, a.bf, false, b.bf,
                                                 (short)0, c, false, false);
}

// ---------------------------------------------------------------------------
// fp32 -> bf16 elementwise
__global__ void cvt_kernel(const float* __restrict__ in,
                           unsigned short* __restrict__ out, int n) {
  int i = blockIdx.x * blockDim.x + threadIdx.x;
  if (i < n) out[i] = f2bf(in[i]);
}

// fp32 [R][C] -> bf16 [C][R] (transpose + convert); one-time, L2-resident
__global__ void transpose_cvt_kernel(const float* __restrict__ in,
                                     unsigned short* __restrict__ out,
                                     int R, int C) {
  int i = blockIdx.x * blockDim.x + threadIdx.x;
  if (i >= R * C) return;
  int r = i / C, c = i - r * C;
  out[(size_t)c * R + r] = f2bf(in[i]);
}

// ---------------------------------------------------------------------------
// QKV projection: one wave computes a 16-row x 128-col (one head) tile.
//   xbf : [B*T][M] bf16 (A rows, K = M)
//   wT  : [H*DHEAD][M] bf16 (B^T rows)
//   out : [B][H][T][DHEAD] bf16, RoPE fused for q/k
__global__ __launch_bounds__(32) void qkv_proj_kernel(
    const unsigned short* __restrict__ xbf,
    const unsigned short* __restrict__ wT,
    unsigned short* __restrict__ outbf,
    int rope_flag) {
  const int lane = threadIdx.x;
  const int r    = lane & 15;
  const int hi   = lane >> 4;
  const int kb   = hi * 8;
  const int row_tile = blockIdx.x;
  const int h        = blockIdx.y;

  v8f acc[8];
#pragma unroll
  for (int n = 0; n < 8; ++n) acc[n] = (v8f)(0.0f);

  const unsigned short* arow  = xbf + (size_t)(row_tile * 16 + r) * M_DIM;
  const unsigned short* bbase = wT + (size_t)(h * DHEAD + r) * M_DIM;

  for (int k0 = 0; k0 < M_DIM; k0 += 32) {
    Frag af;
    af.u[0] = *reinterpret_cast<const uint4*>(arow + k0 + kb);
    af.u[1] = *reinterpret_cast<const uint4*>(arow + k0 + 16 + kb);
#pragma unroll
    for (int n = 0; n < 8; ++n) {
      const unsigned short* brow = bbase + (size_t)(n * 16) * M_DIM;
      Frag bf_;
      bf_.u[0] = *reinterpret_cast<const uint4*>(brow + k0 + kb);
      bf_.u[1] = *reinterpret_cast<const uint4*>(brow + k0 + 16 + kb);
      acc[n] = wmma_bf16(af, bf_, acc[n]);
    }
  }

  if (rope_flag) {
    // acc[n] (n<4) holds even-half dim de = n*16 + r; acc[n+4] holds de + 64.
#pragma unroll
    for (int n = 0; n < 4; ++n) {
      int de = n * 16 + r;
      float freq = __powf(10000.0f, -(float)de / 64.0f);
#pragma unroll
      for (int i = 0; i < 8; ++i) {
        int pos = (row_tile * 16 + i + hi * 8) & (T_DIM - 1);
        float ang = (float)pos * freq;
        float sn, cs;
        __sincosf(ang, &sn, &cs);
        float ev = acc[n][i], od = acc[n + 4][i];
        acc[n][i]     = ev * cs - od * sn;
        acc[n + 4][i] = ev * sn + od * cs;
      }
    }
  }

#pragma unroll
  for (int i = 0; i < 8; ++i) {
    int grow = row_tile * 16 + i + hi * 8;
    int bidx = grow / T_DIM;
    int t    = grow & (T_DIM - 1);
    unsigned short* orow =
        outbf + (((size_t)bidx * H_DIM + h) * T_DIM + t) * DHEAD;
#pragma unroll
    for (int n = 0; n < 8; ++n) orow[n * 16 + r] = f2bf(acc[n][i]);
  }
}

// ---------------------------------------------------------------------------
// Flash attention: block = 4 waves, each wave owns 16 q rows (64 rows/block).
// K tile [32 keys][128] row-major in LDS (async copy on CDNA5);
// V tile transposed [128][32] in LDS.
__global__ __launch_bounds__(128) void attn_kernel(
    const unsigned short* __restrict__ qbf,
    const unsigned short* __restrict__ kbf,
    const unsigned short* __restrict__ vbf,
    unsigned short* __restrict__ obf) {
  __shared__ __align__(16) unsigned short Ks[32 * 128];
  __shared__ __align__(16) unsigned short Vt[128 * 32];
  __shared__ __align__(16) unsigned short Ps[4][16 * 32];

  const int tid  = threadIdx.x;
  const int wave = tid >> 5;
  const int lane = tid & 31;
  const int r    = lane & 15;
  const int hi   = lane >> 4;
  const int kb   = hi * 8;

  const int qt = blockIdx.x;  // 64-row q tile
  const int h  = blockIdx.y;
  const int bz = blockIdx.z;
  const int t0 = qt * 64 + wave * 16;

  const size_t bh = ((size_t)bz * H_DIM + h) * T_DIM;

  // Q fragments for the 4 d-chunks, held in registers for the whole loop.
  Frag qf[4];
  {
    const unsigned short* qrow = qbf + (bh + t0 + r) * DHEAD;
#pragma unroll
    for (int c = 0; c < 4; ++c) {
      qf[c].u[0] = *reinterpret_cast<const uint4*>(qrow + c * 32 + kb);
      qf[c].u[1] = *reinterpret_cast<const uint4*>(qrow + c * 32 + 16 + kb);
    }
  }

  v8f o_acc[8];
#pragma unroll
  for (int n = 0; n < 8; ++n) o_acc[n] = (v8f)(0.0f);
  float m_run[8], l_run[8];
#pragma unroll
  for (int i = 0; i < 8; ++i) { m_run[i] = NEG_INF; l_run[i] = 0.0f; }

  const int jend = qt * 64 + 63;  // last key needed by this block
  for (int j0 = 0; j0 <= jend; j0 += 32) {
    // ---- cooperative K/V tile load: 128 threads, 32 keys x 4 segments ----
    {
      int key  = tid >> 2;
      int dseg = (tid & 3) * 32;
      const unsigned short* kg = kbf + (bh + j0 + key) * DHEAD + dseg;
      unsigned short*       kl = &Ks[key * 128 + dseg];
#if HAVE_ASYNC_LDS
      // 64 B per thread direct global->LDS, tracked with ASYNCcnt (no VGPR
      // round trip, frees the VALU/vmem-return path during the copy).
      gvi4* g = (gvi4*)kg;
      lvi4* l = (lvi4*)kl;
      __builtin_amdgcn_global_load_async_to_lds_b128(g, l, 0, 0);
      __builtin_amdgcn_global_load_async_to_lds_b128(g, l, 16, 0);
      __builtin_amdgcn_global_load_async_to_lds_b128(g, l, 32, 0);
      __builtin_amdgcn_global_load_async_to_lds_b128(g, l, 48, 0);
#else
      const uint4* ksrc = reinterpret_cast<const uint4*>(kg);
      uint4*       kdst = reinterpret_cast<uint4*>(kl);
      kdst[0] = ksrc[0]; kdst[1] = ksrc[1]; kdst[2] = ksrc[2]; kdst[3] = ksrc[3];
#endif
      // V must land transposed in LDS (so PV B-fragments are two b128 ds
      // loads); async copies cannot transpose, so unpack via VGPRs.
      const unsigned int* vsrc = reinterpret_cast<const unsigned int*>(
          vbf + (bh + j0 + key) * DHEAD + dseg);
#pragma unroll
      for (int w = 0; w < 16; ++w) {
        unsigned int pk = vsrc[w];
        Vt[(dseg + 2 * w + 0) * 32 + key] = (unsigned short)(pk & 0xffffu);
        Vt[(dseg + 2 * w + 1) * 32 + key] = (unsigned short)(pk >> 16);
      }
    }
#if HAVE_ASYNC_LDS
    __builtin_amdgcn_s_wait_asynccnt(0);
#endif
    __syncthreads();

    // ---- S = q k^T for two 16-key tiles ----
    v8f s0 = (v8f)(0.0f), s1 = (v8f)(0.0f);
#pragma unroll
    for (int c = 0; c < 4; ++c) {
      Frag b0, b1;
      const unsigned short* k0r = &Ks[r * 128 + c * 32];
      b0.u[0] = *reinterpret_cast<const uint4*>(k0r + kb);
      b0.u[1] = *reinterpret_cast<const uint4*>(k0r + 16 + kb);
      const unsigned short* k1r = &Ks[(16 + r) * 128 + c * 32];
      b1.u[0] = *reinterpret_cast<const uint4*>(k1r + kb);
      b1.u[1] = *reinterpret_cast<const uint4*>(k1r + 16 + kb);
      s0 = wmma_bf16(qf[c], b0, s0);
      s1 = wmma_bf16(qf[c], b1, s1);
    }

    // ---- scale, causal mask, online softmax (row = i + hi*8) ----
    const float sc = 1.0f / (float)DHEAD;
#pragma unroll
    for (int i = 0; i < 8; ++i) {
      int row = t0 + i + hi * 8;
      float e0 = s0[i] * sc; if (j0 + r > row)      e0 = NEG_INF;
      float e1 = s1[i] * sc; if (j0 + 16 + r > row) e1 = NEG_INF;
      float mx = fmaxf(e0, e1);
#pragma unroll
      for (int mm = 1; mm < 16; mm <<= 1) mx = fmaxf(mx, __shfl_xor(mx, mm, 32));
      float mn = fmaxf(m_run[i], mx);
      float alpha = __expf(m_run[i] - mn);
      e0 = __expf(e0 - mn);
      e1 = __expf(e1 - mn);
      float rs = e0 + e1;
#pragma unroll
      for (int mm = 1; mm < 16; mm <<= 1) rs += __shfl_xor(rs, mm, 32);
      l_run[i] = l_run[i] * alpha + rs;
      m_run[i] = mn;
      // stash P (bf16) for layout change C->A
      Ps[wave][(i + hi * 8) * 32 + r]      = f2bf(e0);
      Ps[wave][(i + hi * 8) * 32 + 16 + r] = f2bf(e1);
      // rescale O accumulators for this row
#pragma unroll
      for (int n = 0; n < 8; ++n) o_acc[n][i] *= alpha;
    }
    __syncthreads();

    // ---- O += P V  (A = P [16x32], B = V [32 keys x 16 d] per d-chunk) ----
    Frag pf;
    const unsigned short* prow = &Ps[wave][r * 32];
    pf.u[0] = *reinterpret_cast<const uint4*>(prow + kb);
    pf.u[1] = *reinterpret_cast<const uint4*>(prow + 16 + kb);
#pragma unroll
    for (int n = 0; n < 8; ++n) {
      Frag vf;
      const unsigned short* vrow = &Vt[(n * 16 + r) * 32];
      vf.u[0] = *reinterpret_cast<const uint4*>(vrow + kb);
      vf.u[1] = *reinterpret_cast<const uint4*>(vrow + 16 + kb);
      o_acc[n] = wmma_bf16(pf, vf, o_acc[n]);
    }
    __syncthreads();
  }

  // ---- normalize and emit o as [B][T][H*DHEAD] bf16 for the final GEMM ----
#pragma unroll
  for (int i = 0; i < 8; ++i) {
    float inv = 1.0f / l_run[i];
    int t = t0 + i + hi * 8;
    unsigned short* orow =
        obf + ((size_t)(bz * T_DIM + t)) * (H_DIM * DHEAD) + h * DHEAD;
#pragma unroll
    for (int n = 0; n < 8; ++n) orow[n * 16 + r] = f2bf(o_acc[n][i] * inv);
  }
}

// ---------------------------------------------------------------------------
// Output projection: out[B*T][M] (fp32) = o_bf [B*T][H*D] x wo; woT = [M][H*D]
__global__ __launch_bounds__(32) void out_proj_kernel(
    const unsigned short* __restrict__ obf,
    const unsigned short* __restrict__ woT,
    float* __restrict__ out) {
  const int lane = threadIdx.x;
  const int r  = lane & 15;
  const int hi = lane >> 4;
  const int kb = hi * 8;
  const int row_tile = blockIdx.x;
  const int cg       = blockIdx.y;  // 128-column group

  v8f acc[8];
#pragma unroll
  for (int n = 0; n < 8; ++n) acc[n] = (v8f)(0.0f);

  const unsigned short* arow  = obf + (size_t)(row_tile * 16 + r) * (H_DIM * DHEAD);
  const unsigned short* bbase = woT + (size_t)(cg * 128 + r) * (H_DIM * DHEAD);

  for (int k0 = 0; k0 < H_DIM * DHEAD; k0 += 32) {
    Frag af;
    af.u[0] = *reinterpret_cast<const uint4*>(arow + k0 + kb);
    af.u[1] = *reinterpret_cast<const uint4*>(arow + k0 + 16 + kb);
#pragma unroll
    for (int n = 0; n < 8; ++n) {
      const unsigned short* brow = bbase + (size_t)(n * 16) * (H_DIM * DHEAD);
      Frag bf_;
      bf_.u[0] = *reinterpret_cast<const uint4*>(brow + k0 + kb);
      bf_.u[1] = *reinterpret_cast<const uint4*>(brow + k0 + 16 + kb);
      acc[n] = wmma_bf16(af, bf_, acc[n]);
    }
  }

#pragma unroll
  for (int i = 0; i < 8; ++i) {
    size_t row = (size_t)(row_tile * 16 + i + hi * 8);
#pragma unroll
    for (int n = 0; n < 8; ++n)
      out[row * M_DIM + (cg * 128 + n * 16 + r)] = acc[n][i];
  }
}

// ---------------------------------------------------------------------------
extern "C" void kernel_launch(void* const* d_in, const int* in_sizes, int n_in,
                              void* d_out, int out_size, void* d_ws, size_t ws_size,
                              hipStream_t stream) {
  (void)in_sizes; (void)n_in; (void)out_size; (void)ws_size;

  const float* x  = (const float*)d_in[0];
  const float* wq = (const float*)d_in[1];
  const float* wk = (const float*)d_in[2];
  const float* wv = (const float*)d_in[3];
  const float* wo = (const float*)d_in[4];
  float* out = (float*)d_out;

  const size_t BT = (size_t)B_DIM * T_DIM;       // 4096
  const size_t NX = BT * M_DIM;                  // 8.39M elems
  const size_t NW = (size_t)M_DIM * M_DIM;       // 4.19M elems

  unsigned short* p   = (unsigned short*)d_ws;
  unsigned short* xbf = p;  p += NX;
  unsigned short* wqT = p;  p += NW;
  unsigned short* wkT = p;  p += NW;
  unsigned short* wvT = p;  p += NW;
  unsigned short* woT = p;  p += NW;
  unsigned short* qbf = p;  p += NX;
  unsigned short* kbf = p;  p += NX;
  unsigned short* vbf = p;  p += NX;
  unsigned short* obf = xbf;  // reuse: x_bf dead after the V projection

  const int thr = 256;
  cvt_kernel<<<(unsigned)((NX + thr - 1) / thr), thr, 0, stream>>>(x, xbf, (int)NX);
  transpose_cvt_kernel<<<(unsigned)((NW + thr - 1) / thr), thr, 0, stream>>>(wq, wqT, M_DIM, M_DIM);
  transpose_cvt_kernel<<<(unsigned)((NW + thr - 1) / thr), thr, 0, stream>>>(wk, wkT, M_DIM, M_DIM);
  transpose_cvt_kernel<<<(unsigned)((NW + thr - 1) / thr), thr, 0, stream>>>(wv, wvT, M_DIM, M_DIM);
  transpose_cvt_kernel<<<(unsigned)((NW + thr - 1) / thr), thr, 0, stream>>>(wo, woT, M_DIM, M_DIM);

  dim3 gproj((unsigned)(BT / 16), H_DIM);
  qkv_proj_kernel<<<gproj, 32, 0, stream>>>(xbf, wqT, qbf, 1);
  qkv_proj_kernel<<<gproj, 32, 0, stream>>>(xbf, wkT, kbf, 1);
  qkv_proj_kernel<<<gproj, 32, 0, stream>>>(xbf, wvT, vbf, 0);

  dim3 gattn(T_DIM / 64, H_DIM, B_DIM);
  attn_kernel<<<gattn, 128, 0, stream>>>(qbf, kbf, vbf, obf);

  dim3 gout((unsigned)(BT / 16), M_DIM / 128);
  out_proj_kernel<<<gout, 32, 0, stream>>>(obf, woT, out);
}
